// GAT_39264591020276
// MI455X (gfx1250) — matrix operand order repaired and checked by reading, not compiled, not run on hardware
//
#include <hip/hip_runtime.h>
#include <hip/hip_bf16.h>

// ---------------------------------------------------------------------------
// GAT forward for MI455X (gfx1250, wave32, WMMA).
// B=8 graphs, N=2048 nodes, D=64, H=4 heads, alpha=0.2.
// Rank-1 attention logits: e_ij = lrelu(s1_i + s2_j).
//   - exact softmax max: m_i = lrelu(s1_i + max_j s2_j)
//   - exp-free P tile:   p_ij = max(E1p_i*E2p_j, E1n_i*E2n_j)  (all fp16, <=1)
//   - row sums Z via a 5th WMMA against an all-ones B matrix (f32 accumulate)
// ---------------------------------------------------------------------------

typedef _Float16 v16h __attribute__((ext_vector_type(16)));
typedef _Float16 v8h  __attribute__((ext_vector_type(8)));
typedef float    v8f  __attribute__((ext_vector_type(8)));

#define GAT_B 8
#define GAT_N 2048
#define GAT_D 64
#define GAT_H 4
#define GAT_ALPHA 0.2f

__device__ __forceinline__ v16h cat16(v8h lo, v8h hi) {
  return __builtin_shufflevector(lo, hi, 0,1,2,3,4,5,6,7,8,9,10,11,12,13,14,15);
}

__device__ __forceinline__ v8h pkmax8(v8h x, v8h y) {
#if __has_builtin(__builtin_elementwise_max)
  return __builtin_elementwise_max(x, y);   // v_pk_max_num_f16
#else
  v8h r;
#pragma unroll
  for (int i = 0; i < 8; i++) r[i] = x[i] > y[i] ? x[i] : y[i];
  return r;
#endif
}

__device__ __forceinline__ v8f wmma_f16(v16h a, v16h b, v8f c) {
  // (neg_a, A, neg_b, B, c_mod, C, reuse_a, reuse_b)
  return __builtin_amdgcn_wmma_f32_16x16x32_f16(false, a, false, b,
                                                (short)0, c, false, false);
}

// --------------------------- K0: fp16 prep ---------------------------------
__global__ void prep_kernel(const float* __restrict__ X,
                            const float* __restrict__ W1,
                            const float* __restrict__ W2,
                            _Float16* __restrict__ Xh,
                            _Float16* __restrict__ WT1,
                            _Float16* __restrict__ WT2) {
  int i = blockIdx.x * 256 + threadIdx.x;
  if (i < GAT_B * GAT_N * GAT_D) Xh[i] = (_Float16)X[i];
  if (i < GAT_H * GAT_D * GAT_D) {
    int h = i >> 12, rem = i & 4095, n = rem >> 6, k = rem & 63;
    WT1[i] = (_Float16)W1[(h * 64 + k) * 64 + n];
  }
  if (i < GAT_D * (GAT_H * GAT_D)) {
    int n = i >> 8, k = i & 255;
    WT2[i] = (_Float16)W2[k * 64 + n];
  }
}

// --------------------------- proj GEMM (WMMA) ------------------------------
// OT[seg][64][2048] (f16, transposed) = A[segA][N][K] @ Bt[segB][64][K]^T
__global__ void proj_kernel(const _Float16* __restrict__ A,
                            const _Float16* __restrict__ Bt,
                            _Float16* __restrict__ OT,
                            int K, int hc) {
  const int blocksPerSeg = GAT_N / 128;
  int seg    = blockIdx.x / blocksPerSeg;
  int rowblk = blockIdx.x % blocksPerSeg;
  int segA = seg / hc, segB = seg % hc;
  int wave = threadIdx.x >> 5, lane = threadIdx.x & 31;
  int g = lane >> 4, m = lane & 15;
  int i0 = rowblk * 128 + wave * 16;

  const _Float16* Arow = A + (size_t)segA * GAT_N * K + (size_t)(i0 + m) * K;
  const _Float16* Bseg = Bt + (size_t)segB * 64 * K;

  v8f acc[4] = {};
  for (int kc = 0; kc < K; kc += 32) {
    v16h a = cat16(*(const v8h*)(Arow + kc + 8 * g),
                   *(const v8h*)(Arow + kc + 16 + 8 * g));
#pragma unroll
    for (int nb = 0; nb < 4; nb++) {
      const _Float16* bp = Bseg + (size_t)(nb * 16 + m) * K + kc + 8 * g;
      v16h bv = cat16(*(const v8h*)bp, *(const v8h*)(bp + 16));
      acc[nb] = wmma_f16(a, bv, acc[nb]);
    }
  }
  _Float16* Oseg = OT + (size_t)seg * 64 * GAT_N;
#pragma unroll
  for (int nb = 0; nb < 4; nb++)
#pragma unroll
    for (int r = 0; r < 8; r++) {
      int row = i0 + r + 8 * g;
      int col = nb * 16 + m;
      Oseg[(size_t)col * GAT_N + row] = (_Float16)acc[nb][r];
    }
}

// --------------------------- attention scores ------------------------------
__global__ void scores_kernel(const _Float16* __restrict__ HT,
                              const float* __restrict__ avec,
                              float* __restrict__ s1, float* __restrict__ s2,
                              int aMod, int aStride, int nSeg) {
  int idx = blockIdx.x * blockDim.x + threadIdx.x;
  if (idx >= nSeg * GAT_N) return;
  int seg = idx >> 11, i = idx & (GAT_N - 1);
  const _Float16* Hseg = HT + (size_t)seg * 64 * GAT_N;
  const float* a = avec + (size_t)(seg % aMod) * aStride;
  float a1 = 0.f, a2 = 0.f;
#pragma unroll 8
  for (int d = 0; d < 64; d++) {
    float h = (float)Hseg[(size_t)d * GAT_N + i];
    a1 += h * a[d];
    a2 += h * a[64 + d];
  }
  s1[idx] = a1;
  s2[idx] = a2;
}

// per-segment max over s2
__global__ void rowmax_kernel(const float* __restrict__ s2,
                              float* __restrict__ M2, int len) {
  __shared__ float red[256];
  const float* p = s2 + (size_t)blockIdx.x * len;
  float m = -3.0e38f;
  for (int i = threadIdx.x; i < len; i += 256) m = fmaxf(m, p[i]);
  red[threadIdx.x] = m;
  __syncthreads();
  for (int s = 128; s > 0; s >>= 1) {
    if (threadIdx.x < s)
      red[threadIdx.x] = fmaxf(red[threadIdx.x], red[threadIdx.x + s]);
    __syncthreads();
  }
  if (threadIdx.x == 0) M2[blockIdx.x] = red[0];
}

// E2p[j] = exp(s2_j - M2) <= 1,  E2n[j] = exp(alpha*(s2_j - M2)) <= 1  (fp16)
__global__ void exps_kernel(const float* __restrict__ s2,
                            const float* __restrict__ M2,
                            _Float16* __restrict__ E2p,
                            _Float16* __restrict__ E2n, int nSeg) {
  int idx = blockIdx.x * blockDim.x + threadIdx.x;
  if (idx >= nSeg * GAT_N) return;
  int seg = idx >> 11;
  float d = s2[idx] - M2[seg];
  E2p[idx] = (_Float16)__expf(d);
  E2n[idx] = (_Float16)__expf(GAT_ALPHA * d);
}

// --------------------------- attention aggregation -------------------------
// Each wave owns TWO 16-row tiles (32 rows); E2 and B loads shared across
// tiles. 10 WMMAs per 32-column chunk (2x4 main + 2x1 row-sum vs ones-B).
// MODE 1: elu -> f16 concat buffer.  MODE 2: elu -> log_softmax -> f32 d_out.
template <int MODE>
__global__ void attn_kernel(const float* __restrict__ s1a,
                            const float* __restrict__ M2a,
                            const _Float16* __restrict__ E2pa,
                            const _Float16* __restrict__ E2na,
                            const _Float16* __restrict__ HT,
                            void* __restrict__ outp, int hc) {
  const int blocksPerSeg = GAT_N / 256;
  int seg    = blockIdx.x / blocksPerSeg;
  int rowblk = blockIdx.x % blocksPerSeg;
  int wave = threadIdx.x >> 5, lane = threadIdx.x & 31;
  int g = lane >> 4, m = lane & 15;
  int i0 = rowblk * 256 + wave * 32;

  const float* s1 = s1a + (size_t)seg * GAT_N;
  const _Float16* E2p = E2pa + (size_t)seg * GAT_N;
  const _Float16* E2n = E2na + (size_t)seg * GAT_N;
  const _Float16* Hseg = HT + (size_t)seg * 64 * GAT_N;
  float M2 = M2a[seg];

  // per-row fp16 scalars (all <= 1): E1p = exp(s1+M2-m), E1n = exp(.2(s1+M2)-m)
  v8h e1p8[2], e1n8[2];
#pragma unroll
  for (int t = 0; t < 2; t++) {
    float tt = s1[i0 + t * 16 + m] + M2;
    float mrow = fmaxf(tt, GAT_ALPHA * tt);
    _Float16 e1p = (_Float16)__expf(tt - mrow);
    _Float16 e1n = (_Float16)__expf(GAT_ALPHA * tt - mrow);
    e1p8[t] = (v8h){e1p, e1p, e1p, e1p, e1p, e1p, e1p, e1p};
    e1n8[t] = (v8h){e1n, e1n, e1n, e1n, e1n, e1n, e1n, e1n};
  }

  v16h onesB;
#pragma unroll
  for (int e = 0; e < 16; e++) onesB[e] = (_Float16)1.0f;

  v8f acc[2][4] = {};
  v8f accz[2] = {};

  for (int j0 = 0; j0 < GAT_N; j0 += 32) {
    // E2 factors for this lane's k-runs [8g..8g+7], [16+8g..23+8g]
    v8h p1 = *(const v8h*)(E2p + j0 + 8 * g);
    v8h p2 = *(const v8h*)(E2p + j0 + 16 + 8 * g);
    v8h n1 = *(const v8h*)(E2n + j0 + 8 * g);
    v8h n2 = *(const v8h*)(E2n + j0 + 16 + 8 * g);
    v16h a[2];
#pragma unroll
    for (int t = 0; t < 2; t++)   // p = max(E1p*E2p, E1n*E2n), packed fp16
      a[t] = cat16(pkmax8(p1 * e1p8[t], n1 * e1n8[t]),
                   pkmax8(p2 * e1p8[t], n2 * e1n8[t]));
#pragma unroll
    for (int nb = 0; nb < 4; nb++) {
      const _Float16* bp = Hseg + (size_t)(nb * 16 + m) * GAT_N + j0 + 8 * g;
      if (j0 + 32 < GAT_N) __builtin_prefetch(bp + 32, 0, 1);
      v16h bv = cat16(*(const v8h*)bp, *(const v8h*)(bp + 16));
      acc[0][nb] = wmma_f16(a[0], bv, acc[0][nb]);
      acc[1][nb] = wmma_f16(a[1], bv, acc[1][nb]);
    }
    accz[0] = wmma_f16(a[0], onesB, accz[0]);   // Z row-sums, f32 accumulate
    accz[1] = wmma_f16(a[1], onesB, accz[1]);
  }

  // accz[t][r] holds Z for row (i0 + t*16 + r + 8g) in every column lane.
  if (MODE == 1) {
    _Float16* xc = (_Float16*)outp;
    int b = seg / hc, h = seg % hc;
#pragma unroll
    for (int t = 0; t < 2; t++)
#pragma unroll
      for (int r = 0; r < 8; r++) {
        float Zr = accz[t][r];
        int row = i0 + t * 16 + r + 8 * g;
#pragma unroll
        for (int nb = 0; nb < 4; nb++) {
          float v = acc[t][nb][r] / Zr;
          float e = v > 0.f ? v : (__expf(v) - 1.f);   // elu
          xc[((size_t)(b * GAT_N + row) * (GAT_H * GAT_D)) + h * 64 + nb * 16 + m] =
              (_Float16)e;
        }
      }
  } else {
    float* out = (float*)outp;
#pragma unroll
    for (int t = 0; t < 2; t++)
#pragma unroll
      for (int r = 0; r < 8; r++) {
        float Zr = accz[t][r];
        int row = i0 + t * 16 + r + 8 * g;
        float v[4], pm = -3.0e38f;
#pragma unroll
        for (int nb = 0; nb < 4; nb++) {
          float x = acc[t][nb][r] / Zr;
          v[nb] = x > 0.f ? x : (__expf(x) - 1.f);     // elu
          pm = fmaxf(pm, v[nb]);
        }
#pragma unroll
        for (int off = 1; off < 16; off <<= 1) pm = fmaxf(pm, __shfl_xor(pm, off, 32));
        float se = 0.f;
#pragma unroll
        for (int nb = 0; nb < 4; nb++) se += __expf(v[nb] - pm);
#pragma unroll
        for (int off = 1; off < 16; off <<= 1) se += __shfl_xor(se, off, 32);
        float lse = pm + __logf(se);
#pragma unroll
        for (int nb = 0; nb < 4; nb++)
          out[(size_t)(seg * GAT_N + row) * 64 + nb * 16 + m] = v[nb] - lse;
      }
  }
}

// ---------------------------------------------------------------------------
extern "C" void kernel_launch(void* const* d_in, const int* in_sizes, int n_in,
                              void* d_out, int out_size, void* d_ws, size_t ws_size,
                              hipStream_t stream) {
  const float* h_states = (const float*)d_in[0];   // [B*N, 64]
  const float* W_heads  = (const float*)d_in[1];   // [H, 64, 64]
  const float* a_heads  = (const float*)d_in[2];   // [H, 128]
  const float* W_out    = (const float*)d_in[3];   // [256, 64]
  const float* a_out    = (const float*)d_in[4];   // [128]
  // d_in[5] seq_start_end: equal contiguous segments (baked into indexing)

  char* ws = (char*)d_ws;
  size_t off = 0;
  auto take = [&](size_t bytes) -> void* {
    void* p = ws + off;
    off = (off + bytes + 255) & ~(size_t)255;
    return p;
  };
  _Float16* Xh    = (_Float16*)take((size_t)GAT_B * GAT_N * GAT_D * 2);
  _Float16* WT1   = (_Float16*)take((size_t)GAT_H * GAT_D * GAT_D * 2);
  _Float16* WT2   = (_Float16*)take((size_t)GAT_D * GAT_H * GAT_D * 2);
  _Float16* HT1   = (_Float16*)take((size_t)GAT_B * GAT_H * GAT_D * GAT_N * 2);
  float*    s1_1  = (float*)take((size_t)GAT_B * GAT_H * GAT_N * 4);
  float*    s2_1  = (float*)take((size_t)GAT_B * GAT_H * GAT_N * 4);
  float*    M2_1  = (float*)take((size_t)GAT_B * GAT_H * 4);
  _Float16* E2p_1 = (_Float16*)take((size_t)GAT_B * GAT_H * GAT_N * 2);
  _Float16* E2n_1 = (_Float16*)take((size_t)GAT_B * GAT_H * GAT_N * 2);
  _Float16* XC    = (_Float16*)take((size_t)GAT_B * GAT_N * GAT_H * GAT_D * 2);
  _Float16* HT2   = (_Float16*)take((size_t)GAT_B * GAT_D * GAT_N * 2);
  float*    s1_2  = (float*)take((size_t)GAT_B * GAT_N * 4);
  float*    s2_2  = (float*)take((size_t)GAT_B * GAT_N * 4);
  float*    M2_2  = (float*)take((size_t)GAT_B * 4);
  _Float16* E2p_2 = (_Float16*)take((size_t)GAT_B * GAT_N * 2);
  _Float16* E2n_2 = (_Float16*)take((size_t)GAT_B * GAT_N * 2);
  (void)ws_size; (void)in_sizes; (void)n_in; (void)out_size;

  // K0: fp16 conversions / weight transposes
  prep_kernel<<<(GAT_B * GAT_N * GAT_D + 255) / 256, 256, 0, stream>>>(
      h_states, W_heads, W_out, Xh, WT1, WT2);

  // Layer 1
  proj_kernel<<<GAT_B * GAT_H * (GAT_N / 128), 256, 0, stream>>>(
      Xh, WT1, HT1, GAT_D, GAT_H);
  scores_kernel<<<(GAT_B * GAT_H * GAT_N) / 256, 256, 0, stream>>>(
      HT1, a_heads, s1_1, s2_1, GAT_H, 2 * GAT_D, GAT_B * GAT_H);
  rowmax_kernel<<<GAT_B * GAT_H, 256, 0, stream>>>(s2_1, M2_1, GAT_N);
  exps_kernel<<<(GAT_B * GAT_H * GAT_N) / 256, 256, 0, stream>>>(
      s2_1, M2_1, E2p_1, E2n_1, GAT_B * GAT_H);
  attn_kernel<1><<<GAT_B * GAT_H * (GAT_N / 256), 256, 0, stream>>>(
      s1_1, M2_1, E2p_1, E2n_1, HT1, (void*)XC, GAT_H);

  // Layer 2
  proj_kernel<<<GAT_B * (GAT_N / 128), 256, 0, stream>>>(
      XC, WT2, HT2, GAT_H * GAT_D, 1);
  scores_kernel<<<(GAT_B * GAT_N) / 256, 256, 0, stream>>>(
      HT2, a_out, s1_2, s2_2, 1, 0, GAT_B);
  rowmax_kernel<<<GAT_B, 256, 0, stream>>>(s2_2, M2_2, GAT_N);
  exps_kernel<<<(GAT_B * GAT_N) / 256, 256, 0, stream>>>(
      s2_2, M2_2, E2p_2, E2n_2, GAT_B);
  attn_kernel<2><<<GAT_B * (GAT_N / 256), 256, 0, stream>>>(
      s1_2, M2_2, E2p_2, E2n_2, HT2, d_out, 1);
}
